// GCAModel_26800595927371
// MI455X (gfx1250) — matrix operand order repaired and checked by reading, not compiled, not run on hardware
//
#include <hip/hip_runtime.h>

#define B_   16
#define CDD_ 5
#define HIS_ 50
#define S_   20
#define T_   41
#define E_   300
#define F_   256
#define H_   16
#define DV_  16
#define QD_  200

#define TP   48      // padded T (3 M-tiles)
#define SP   64      // padded s-dim for attn K (2 K-steps of 32)
#define EP   320     // padded E (so 3*EP = 960 is a multiple of 32)
#define KC   960     // conv GEMM K
#define QDP  208     // padded query dim (13 N-tiles)

#define SCALE_ 0.05773502691896258f   // 1/sqrt(300)

typedef __attribute__((ext_vector_type(16))) __bf16 bf16x16;
typedef __attribute__((ext_vector_type(8)))  __bf16 bf16x8;
typedef __attribute__((ext_vector_type(8)))  float  f32x8;

union BF16X16 { bf16x16 v; bf16x8 h[2]; __bf16 e[16]; };

static __device__ inline f32x8 wmma_bf16(bf16x16 a, bf16x16 b, f32x8 c) {
  return __builtin_amdgcn_wmma_f32_16x16x32_bf16(false, a, false, b, (short)0, c, false, false);
}

// A fragment (16x32) from row-major [M x K] bf16, tile origin (mbase, kbase).
// lane<16: rows m=lane, K = kbase+{0..7} and kbase+{16..23}
// lane>=16: rows m=lane-16, K = kbase+{8..15} and kbase+{24..31}
static __device__ inline bf16x16 lda(const __bf16* buf, int mbase, int kbase, int stride, int lane) {
  const __bf16* p = buf + (mbase + (lane & 15)) * stride + kbase + ((lane >> 4) << 3);
  BF16X16 u;
  u.h[0] = *(const bf16x8*)p;
  u.h[1] = *(const bf16x8*)(p + 16);
  return u.v;
}

// B fragment (32x16) from B^T stored row-major [N x K] bf16.
// lane<16: col n=lane, K = kbase+{0..7} and kbase+{16..23}; lane>=16: col n=lane-16, K = kbase+{8..15} and kbase+{24..31}
static __device__ inline bf16x16 ldb(const __bf16* bt, int nbase, int kbase, int stride, int lane) {
  const __bf16* p = bt + (nbase + (lane & 15)) * stride + kbase + ((lane >> 4) << 3);
  BF16X16 u;
  u.h[0] = *(const bf16x8*)p;
  u.h[1] = *(const bf16x8*)(p + 16);
  return u.v;
}

// A fragment from f32 row-major buffer, converted to bf16 on the fly.
static __device__ inline bf16x16 lda_f32(const float* buf, int mbase, int kbase, int stride, int lane) {
  const float* p = buf + (mbase + (lane & 15)) * stride + kbase + ((lane >> 4) << 3);
  BF16X16 u;
#pragma unroll
  for (int i = 0; i < 8; ++i) u.e[i] = (__bf16)p[i];
#pragma unroll
  for (int i = 0; i < 8; ++i) u.e[8 + i] = (__bf16)p[16 + i];
  return u.v;
}

// ---------------- prep kernels: f32 -> bf16, transposed/padded weight layouts ----------------

__global__ __launch_bounds__(256) void prep_emb(const float* __restrict__ emb,
                                                __bf16* __restrict__ embB) {
  long long i = (long long)blockIdx.x * 256 + threadIdx.x;   // 50000*320
  if (i >= 50000LL * EP) return;
  int col = (int)(i % EP); long long row = i / EP;
  embB[i] = (col < E_) ? (__bf16)emb[row * E_ + col] : (__bf16)0.0f;
}

__global__ __launch_bounds__(256) void prep_w(const float* __restrict__ conv_w,
                                              const float* __restrict__ Wq,
                                              const float* __restrict__ Wv,
                                              const float* __restrict__ Wk,
                                              __bf16* __restrict__ WcT,
                                              __bf16* __restrict__ WqT,
                                              __bf16* __restrict__ WvT,
                                              __bf16* __restrict__ WkT) {
  int i = blockIdx.x * 256 + threadIdx.x;
  if (i < F_ * KC) {                 // WcT[f][k], k = j*EP + e ; = conv_w[j][e][f]
    int f = i / KC, k = i % KC;
    int j = k / EP, e = k % EP;
    WcT[i] = (e < E_) ? (__bf16)conv_w[(j * E_ + e) * F_ + f] : (__bf16)0.0f;
    return;
  }
  i -= F_ * KC;
  if (i < H_ * F_ * F_) {            // WqT[h][g][f] = Wq[h][f][g]
    int h = i >> 16, rem = i & 65535, g = rem >> 8, f = rem & 255;
    WqT[i] = (__bf16)Wq[(h * F_ + f) * F_ + g];
    return;
  }
  i -= H_ * F_ * F_;
  if (i < H_ * DV_ * F_) {           // WvT[h][d][f] = Wv[h][f][d]
    int h = i >> 12, rem = i & 4095, d = rem >> 8, f = rem & 255;
    WvT[i] = (__bf16)Wv[(h * F_ + f) * DV_ + d];
    return;
  }
  i -= H_ * DV_ * F_;
  if (i < QDP * F_) {                // WkT[n][f] = Wk[f][n], zero rows n>=200
    int n = i >> 8, f = i & 255;
    WkT[i] = (n < QD_) ? (__bf16)Wk[f * QD_ + n] : (__bf16)0.0f;
  }
}

// ---------------- fused per-sequence kernel: embed -> conv -> MHSA -> word attention ----------------

__global__ __launch_bounds__(256) void gca_fused(
    const int* __restrict__ cand, const int* __restrict__ click,
    const __bf16* __restrict__ embB, const __bf16* __restrict__ WcT,
    const float* __restrict__ conv_b,
    const __bf16* __restrict__ WqT, const float* __restrict__ bq,
    const __bf16* __restrict__ WvT, const float* __restrict__ bv,
    const __bf16* __restrict__ WkT, const float* __restrict__ bk,
    const float* __restrict__ qw, float* __restrict__ rep_out)
{
  __shared__ __bf16 sh_x[TP * F_];     // x (row-major), 24.0 KB
  __shared__ __bf16 sh_xT[F_ * SP];    // x^T, s padded to 64, 32.0 KB
  __shared__ __bf16 sh_q[TP * F_];     // q, reused as attn (bf16), 24.0 KB
  __shared__ float  sh_sc[TP * TP];    // scores f32, 9.0 KB
  __shared__ __bf16 sh_p[TP * SP];     // softmax probs bf16, 6.0 KB
  __shared__ float  sh_val[TP * F_];   // concat-head values f32, 48 KB (aliases emb buffer)
  __shared__ float  sh_kq[64];

  const int tid  = threadIdx.x;
  const int lane = tid & 31;
  const int wave = tid >> 5;
  const int seq  = blockIdx.x;
  const int b    = seq / (CDD_ * HIS_);
  const int c    = (seq / HIS_) % CDD_;
  const int h    = seq % HIS_;

  // ---- phase 0: gather token embeddings into sliding-window buffer (50 rows x 320 bf16) ----
  __bf16* embF = (__bf16*)sh_val;                   // 50*320*2 = 32000 B <= 49152 B
  unsigned int* embFu = (unsigned int*)embF;
  for (int i = tid; i < 50 * (EP / 2); i += 256) embFu[i] = 0u;
  // zero the s-padding columns of x^T (m = 48..63)
  for (int i = tid; i < F_ * 16; i += 256) sh_xT[(i >> 4) * SP + 48 + (i & 15)] = (__bf16)0.0f;
  __syncthreads();
  for (int i = tid; i < T_ * (EP / 2); i += 256) {
    int t = i / (EP / 2), col = i % (EP / 2);
    int tok = (t < S_) ? cand[(b * CDD_ + c) * S_ + t]
                       : ((t == S_) ? 1 : click[(b * HIS_ + h) * S_ + (t - S_ - 1)]);
    embFu[(t + 1) * (EP / 2) + col] = ((const unsigned int*)embB)[tok * (EP / 2) + col];
  }
  __syncthreads();

  // ---- phase 1: Conv1d(k=3) as GEMM [48 x 960] @ [960 x 256], +bias, ReLU ----
#pragma unroll 1
  for (int p = wave; p < 24; p += 8) {
    int mt = p >> 3, nt0 = (p & 7) * 2;
    f32x8 acc0 = {0.f,0.f,0.f,0.f,0.f,0.f,0.f,0.f};
    f32x8 acc1 = acc0;
#pragma unroll 2
    for (int kt = 0; kt < KC / 32; ++kt) {
      bf16x16 a = lda(embF, mt * 16, kt * 32, EP, lane);   // rows overlap: im2col window
      acc0 = wmma_bf16(a, ldb(WcT, nt0 * 16,      kt * 32, KC, lane), acc0);
      acc1 = wmma_bf16(a, ldb(WcT, nt0 * 16 + 16, kt * 32, KC, lane), acc1);
    }
    int n0 = nt0 * 16 + (lane & 15);
    float b0 = conv_b[n0], b1 = conv_b[n0 + 16];
#pragma unroll
    for (int r = 0; r < 8; ++r) {
      int m = mt * 16 + r + ((lane >> 4) << 3);
      __bf16 x0 = (__bf16)fmaxf(acc0[r] + b0, 0.f);
      __bf16 x1 = (__bf16)fmaxf(acc1[r] + b1, 0.f);
      sh_x[m * F_ + n0]            = x0;
      sh_x[m * F_ + n0 + 16]       = x1;
      sh_xT[n0 * SP + m]           = x0;
      sh_xT[(n0 + 16) * SP + m]    = x1;
    }
  }
  __syncthreads();   // embF (aliased with sh_val) dead; sh_val fully rewritten by heads

  // ---- phase 2: 16 heads, all 8 waves cooperate per head ----
#pragma unroll 1
  for (int hd = 0; hd < H_; ++hd) {
    const __bf16* Wqh = WqT + hd * F_ * F_;
    // 2a: q = x @ Wq_h + bq_h  [48 x 256]
#pragma unroll 1
    for (int p = wave; p < 24; p += 8) {
      int mt = p >> 3, nt0 = (p & 7) * 2;
      f32x8 acc0 = {0.f,0.f,0.f,0.f,0.f,0.f,0.f,0.f};
      f32x8 acc1 = acc0;
#pragma unroll 2
      for (int kt = 0; kt < 8; ++kt) {
        bf16x16 a = lda(sh_x, mt * 16, kt * 32, F_, lane);
        acc0 = wmma_bf16(a, ldb(Wqh, nt0 * 16,      kt * 32, F_, lane), acc0);
        acc1 = wmma_bf16(a, ldb(Wqh, nt0 * 16 + 16, kt * 32, F_, lane), acc1);
      }
      int n0 = nt0 * 16 + (lane & 15);
      float b0 = bq[hd * F_ + n0], b1 = bq[hd * F_ + n0 + 16];
#pragma unroll
      for (int r = 0; r < 8; ++r) {
        int m = mt * 16 + r + ((lane >> 4) << 3);
        sh_q[m * F_ + n0]      = (__bf16)(acc0[r] + b0);
        sh_q[m * F_ + n0 + 16] = (__bf16)(acc1[r] + b1);
      }
    }
    __syncthreads();
    // 2b: scores = q @ x^T * SCALE  [48 x 48]
#pragma unroll 1
    for (int tile = wave; tile < 9; tile += 8) {
      int mt = tile / 3, st = tile % 3;
      f32x8 acc = {0.f,0.f,0.f,0.f,0.f,0.f,0.f,0.f};
#pragma unroll 2
      for (int kt = 0; kt < 8; ++kt)
        acc = wmma_bf16(lda(sh_q, mt * 16, kt * 32, F_, lane),
                        ldb(sh_x, st * 16, kt * 32, F_, lane), acc);
      int n = st * 16 + (lane & 15);
#pragma unroll
      for (int r = 0; r < 8; ++r)
        sh_sc[(mt * 16 + r + ((lane >> 4) << 3)) * TP + n] = acc[r] * SCALE_;
    }
    __syncthreads();
    // 2c: row softmax over s<41, probs -> bf16 (padded cols zeroed)
    if (tid < TP) {
      float mx = -1e30f;
      for (int s = 0; s < T_; ++s) mx = fmaxf(mx, sh_sc[tid * TP + s]);
      float sum = 0.f;
      for (int s = 0; s < T_; ++s) { float e = __expf(sh_sc[tid * TP + s] - mx); sh_sc[tid * TP + s] = e; sum += e; }
      float inv = 1.f / sum;
      for (int s = 0; s < T_; ++s)  sh_p[tid * SP + s] = (__bf16)(sh_sc[tid * TP + s] * inv);
      for (int s = T_; s < SP; ++s) sh_p[tid * SP + s] = (__bf16)0.0f;
    }
    __syncthreads();
    // 2d: attn = P @ x  [48 x 256] -> bf16 into sh_q (q is dead)
#pragma unroll 1
    for (int p = wave; p < 24; p += 8) {
      int mt = p >> 3, nt0 = (p & 7) * 2;
      f32x8 acc0 = {0.f,0.f,0.f,0.f,0.f,0.f,0.f,0.f};
      f32x8 acc1 = acc0;
#pragma unroll 2
      for (int kt = 0; kt < 2; ++kt) {
        bf16x16 a = lda(sh_p, mt * 16, kt * 32, SP, lane);
        acc0 = wmma_bf16(a, ldb(sh_xT, nt0 * 16,      kt * 32, SP, lane), acc0);
        acc1 = wmma_bf16(a, ldb(sh_xT, nt0 * 16 + 16, kt * 32, SP, lane), acc1);
      }
      int n0 = nt0 * 16 + (lane & 15);
#pragma unroll
      for (int r = 0; r < 8; ++r) {
        int m = mt * 16 + r + ((lane >> 4) << 3);
        sh_q[m * F_ + n0]      = (__bf16)acc0[r];
        sh_q[m * F_ + n0 + 16] = (__bf16)acc1[r];
      }
    }
    __syncthreads();
    // 2e: v = attn @ Wv_h + bv_h -> val columns [hd*16, hd*16+16)
    if (wave < 3) {
      const __bf16* Wvh = WvT + hd * DV_ * F_;
      int mt = wave;
      f32x8 acc = {0.f,0.f,0.f,0.f,0.f,0.f,0.f,0.f};
#pragma unroll 2
      for (int kt = 0; kt < 8; ++kt)
        acc = wmma_bf16(lda(sh_q, mt * 16, kt * 32, F_, lane),
                        ldb(Wvh, 0, kt * 32, F_, lane), acc);
      int d = lane & 15;
      float bias = bv[hd * DV_ + d];
#pragma unroll
      for (int r = 0; r < 8; ++r) {
        int m = mt * 16 + r + ((lane >> 4) << 3);
        sh_val[m * F_ + hd * DV_ + d] = acc[r] + bias;
      }
    }
    __syncthreads();
  }

  // ---- phase 3: additive word attention: kq[t] = tanh(val@Wk + bk) . q_words ----
  if (tid < 64) sh_kq[tid] = 0.f;
  __syncthreads();
#pragma unroll 1
  for (int tile = wave; tile < 3 * (QDP / 16); tile += 8) {
    int mt = tile / (QDP / 16), nt = tile % (QDP / 16);
    f32x8 acc = {0.f,0.f,0.f,0.f,0.f,0.f,0.f,0.f};
#pragma unroll 2
    for (int kt = 0; kt < 8; ++kt)
      acc = wmma_bf16(lda_f32(sh_val, mt * 16, kt * 32, F_, lane),
                      ldb(WkT, nt * 16, kt * 32, F_, lane), acc);
    int n = nt * 16 + (lane & 15);
    float bkv = (n < QD_) ? bk[n] : 0.f;
    float qv  = (n < QD_) ? qw[n] : 0.f;
#pragma unroll
    for (int r = 0; r < 8; ++r) {
      float contrib = tanhf(acc[r] + bkv) * qv;
      contrib += __shfl_xor(contrib, 1);
      contrib += __shfl_xor(contrib, 2);
      contrib += __shfl_xor(contrib, 4);
      contrib += __shfl_xor(contrib, 8);
      if ((lane & 15) == 0)
        atomicAdd(&sh_kq[mt * 16 + r + ((lane >> 4) << 3)], contrib);
    }
  }
  __syncthreads();
  if (tid == 0) {       // word softmax over t<41
    float mx = -1e30f;
    for (int t = 0; t < T_; ++t) { float s = sh_kq[t] * SCALE_; sh_kq[t] = s; mx = fmaxf(mx, s); }
    float sum = 0.f;
    for (int t = 0; t < T_; ++t) { float e = __expf(sh_kq[t] - mx); sh_kq[t] = e; sum += e; }
    float inv = 1.f / sum;
    for (int t = 0; t < T_; ++t) sh_kq[t] *= inv;
  }
  __syncthreads();
  {   // rep[f] = sum_t w[t] * val[t][f]
    float r = 0.f;
    for (int t = 0; t < T_; ++t) r += sh_kq[t] * sh_val[t * F_ + tid];
    rep_out[seq * F_ + tid] = r;
  }
}

// ---------------- history mean + final scoring ----------------

__global__ __launch_bounds__(256) void reduce_his(const float* __restrict__ rep,
                                                  float* __restrict__ fv) {
  int bc = blockIdx.x, f = threadIdx.x;
  const float* p = rep + (long long)bc * HIS_ * F_ + f;
  float s = 0.f;
  for (int h2 = 0; h2 < HIS_; ++h2) s += p[h2 * F_];
  fv[bc * F_ + f] = s * (1.0f / HIS_);
}

__global__ __launch_bounds__(128) void final_score(const float* __restrict__ fv,
                                                   const float* __restrict__ Wl,
                                                   const float* __restrict__ bl,
                                                   float* __restrict__ out) {
  __shared__ float sc[B_ * CDD_];
  int tid = threadIdx.x;
  if (tid < B_ * CDD_) {
    float s = 0.f;
    for (int f = 0; f < F_; ++f) s += fv[tid * F_ + f] * Wl[f];
    sc[tid] = s + bl[0];
  }
  __syncthreads();
  if (tid < B_) {
    float mx = -1e30f;
    for (int cc = 0; cc < CDD_; ++cc) mx = fmaxf(mx, sc[tid * CDD_ + cc]);
    float sum = 0.f;
    for (int cc = 0; cc < CDD_; ++cc) sum += __expf(sc[tid * CDD_ + cc] - mx);
    float l = __logf(sum) + mx;
    for (int cc = 0; cc < CDD_; ++cc) out[tid * CDD_ + cc] = sc[tid * CDD_ + cc] - l;
  }
}

// ---------------- launcher ----------------

extern "C" void kernel_launch(void* const* d_in, const int* in_sizes, int n_in,
                              void* d_out, int out_size, void* d_ws, size_t ws_size,
                              hipStream_t stream) {
  (void)in_sizes; (void)n_in; (void)out_size; (void)ws_size;
  const int*   cand   = (const int*)d_in[0];
  const int*   click  = (const int*)d_in[1];
  const float* emb    = (const float*)d_in[2];
  const float* conv_w = (const float*)d_in[3];
  const float* conv_b = (const float*)d_in[4];
  const float* Wq     = (const float*)d_in[5];
  const float* bq     = (const float*)d_in[6];
  const float* Wv     = (const float*)d_in[7];
  const float* bv     = (const float*)d_in[8];
  const float* Wk     = (const float*)d_in[9];
  const float* bk     = (const float*)d_in[10];
  const float* qwv    = (const float*)d_in[11];
  const float* Wl     = (const float*)d_in[12];
  const float* bl     = (const float*)d_in[13];

  char* ws = (char*)d_ws;
  const size_t OFF_EMB = 0;                                  // 50000*320*2 = 32,000,000
  const size_t OFF_WC  = OFF_EMB + 50000ULL * EP * 2;        // 256*960*2   =    491,520
  const size_t OFF_WQ  = OFF_WC  + (size_t)F_ * KC * 2;      // 16*256*256*2 = 2,097,152
  const size_t OFF_WV  = OFF_WQ  + (size_t)H_ * F_ * F_ * 2; // 16*16*256*2 =    131,072
  const size_t OFF_WK  = OFF_WV  + (size_t)H_ * DV_ * F_ * 2;// 208*256*2   =    106,496
  const size_t OFF_REP = OFF_WK  + (size_t)QDP * F_ * 2;     // 4000*256*4  =  4,096,000
  const size_t OFF_FV  = OFF_REP + 4000ULL * F_ * 4;         // 80*256*4    =     81,920

  __bf16* embB = (__bf16*)(ws + OFF_EMB);
  __bf16* WcT  = (__bf16*)(ws + OFF_WC);
  __bf16* WqT  = (__bf16*)(ws + OFF_WQ);
  __bf16* WvT  = (__bf16*)(ws + OFF_WV);
  __bf16* WkT  = (__bf16*)(ws + OFF_WK);
  float*  repb = (float*)(ws + OFF_REP);
  float*  fv   = (float*)(ws + OFF_FV);

  prep_emb<<<(50000 * EP + 255) / 256, 256, 0, stream>>>(emb, embB);
  prep_w<<<(F_ * KC + H_ * F_ * F_ + H_ * DV_ * F_ + QDP * F_ + 255) / 256, 256, 0, stream>>>(
      conv_w, Wq, Wv, Wk, WcT, WqT, WvT, WkT);
  gca_fused<<<B_ * CDD_ * HIS_, 256, 0, stream>>>(
      cand, click, embB, WcT, conv_b, WqT, bq, WvT, bv, WkT, bk, qwv, repb);
  reduce_his<<<B_ * CDD_, 256, 0, stream>>>(repb, fv);
  final_score<<<1, 128, 0, stream>>>(fv, Wl, bl, (float*)d_out);
}
